// EncoderViTRoPE_84138409329215
// MI455X (gfx1250) — matrix operand order repaired and checked by reading, not compiled, not run on hardware
//
#include <hip/hip_runtime.h>
#include <hip/hip_bf16.h>

// ---------------- model constants ----------------
#define BATCH   4
#define NPATCH  512          // 8*64 patches
#define NMETA   3
#define LSEQ    515          // NMETA + NPATCH
#define LP      576          // padded seq: multiple of 64 (and 32 for K)
#define DMODEL  384
#define NHEADS  6
#define HD      64
#define FF      1536
#define DEPTH   12
#define ROWS_ALL (BATCH*LP)  // 2304

typedef __attribute__((ext_vector_type(16))) __bf16 v16bf;
typedef __attribute__((ext_vector_type(8)))  float  v8f;

union BfVec { v16bf v; unsigned short u[16]; unsigned int w[8]; };

__device__ __forceinline__ unsigned short f2bf(float f) {
    unsigned int x = __float_as_uint(f);
    unsigned int r = (x + 0x7fffu + ((x >> 16) & 1u)) >> 16;   // RNE
    return (unsigned short)r;
}
__device__ __forceinline__ unsigned int pack2bf(float lo, float hi) {
    return (unsigned int)f2bf(lo) | ((unsigned int)f2bf(hi) << 16);
}

__device__ __forceinline__ void epilogue(float* __restrict__ Cb, const v8f& acc,
                                         int cmBase, int cn, int ldc,
                                         const float* __restrict__ bias,
                                         const float* __restrict__ gamma, int mode)
{
    const float bval = bias ? bias[cn] : 0.0f;
    const float gval = (mode == 1) ? gamma[cn] : 0.0f;
#pragma unroll
    for (int i = 0; i < 8; ++i) {
        float v = acc[i] + bval;
        if (mode == 2) v = 0.5f * v * (1.0f + erff(v * 0.70710678118654752f));
        const long long ci = (long long)(cmBase + i) * ldc + cn;
        if (mode == 1) v = Cb[ci] + gval * v;
        Cb[ci] = v;
    }
}

// -------------------------------------------------------------------------
// LDS-staged bf16 WMMA GEMM.
// Block = 128 threads (4 waves) -> 64x64 C tile; each wave a 32x32 quadrant
// via 4 x v_wmma_f32_16x16x32_bf16 per 32-wide K step.
// Per K step the block stages A(64x32) and B(32x64) into LDS as bf16:
//   lds_a : [m(64)][k(32)]  bf16   (A rows contiguous in K)
//   lds_b : [n(64)][k(32)]  bf16   (B cols contiguous in K)
// so all fragment reads are contiguous LDS loads (no strided pointer blowup).
//   A: f32 [rows x K] row-major (lda, batch/head strides aB/aH)
//   B: f32 [K x N] (bTrans=0) or [N x K] (bTrans=1)
//   mode 0: C = acc + bias
//   mode 1: C = C + gamma[n] * (acc + bias)   (residual, in place)
//   mode 2: C = gelu(acc + bias)
// grid.x = N/64, grid.y = M/64, grid.z = batch*heads
// -------------------------------------------------------------------------
__global__ void __launch_bounds__(128)
gemm_wmma(const float* __restrict__ A, int lda, long long aB, long long aH,
          const float* __restrict__ Bm, int ldb, long long bB, long long bH, int bTrans,
          float* __restrict__ C, int ldc, long long cB, long long cH,
          const float* __restrict__ bias, const float* __restrict__ gamma,
          int mode, int rows, int K, int nheads)
{
    __shared__ unsigned short lds_a16[64 * 32];
    __shared__ unsigned short lds_b16[64 * 32];

    const int t    = threadIdx.x;
    const int wv   = t >> 5;            // wave 0..3
    const int lane = t & 31;
    const int nt = blockIdx.x, mt = blockIdx.y, z = blockIdx.z;
    const int b  = z / nheads, hh = z % nheads;

    const float* Ab = A  + (long long)b * aB + (long long)hh * aH;
    const float* Bb = Bm + (long long)b * bB + (long long)hh * bH;
    float*       Cb = C  + (long long)b * cB + (long long)hh * cH;

    const int mBase = mt * 64;
    const int nBase = nt * 64;
    const int wm = (wv >> 1) * 32;      // wave's row quadrant within tile
    const int wn = (wv & 1) * 32;       // wave's col quadrant within tile

    // staging thread roles
    const int sRow  = t >> 1;           // 0..63 (A rows / B-trans N rows)
    const int sHalf = t & 1;            // 16-float half
    const int sKrow = t >> 2;           // 0..31 (B non-trans K rows)
    const int sQ    = t & 3;            // 16-float quarter along N

    // fragment roles
    const int l15 = lane & 15;
    const int kh  = lane >> 4;          // 0/1 : K sub-block select
    const int mloc = wm + l15;
    const int nloc = wn + l15;

    v8f acc00, acc01, acc10, acc11;
#pragma unroll
    for (int i = 0; i < 8; ++i) { acc00[i] = 0.f; acc01[i] = 0.f; acc10[i] = 0.f; acc11[i] = 0.f; }

    const unsigned int* au = (const unsigned int*)lds_a16;
    const unsigned int* bu = (const unsigned int*)lds_b16;

    for (int k0 = 0; k0 < K; k0 += 32) {
        __syncthreads();   // previous iteration's fragment reads done

        // ---- stage A: 64 rows x 32 K, 2 threads/row, 16 floats each ----
        {
            unsigned int* dst = (unsigned int*)lds_a16 + sRow * 16 + sHalf * 8;
            const int gr = mBase + sRow;
            if (gr < rows) {
                const float4* ap = (const float4*)(Ab + (long long)gr * lda + k0 + sHalf * 16);
                float4 f0 = ap[0], f1 = ap[1], f2 = ap[2], f3 = ap[3];
                dst[0] = pack2bf(f0.x, f0.y); dst[1] = pack2bf(f0.z, f0.w);
                dst[2] = pack2bf(f1.x, f1.y); dst[3] = pack2bf(f1.z, f1.w);
                dst[4] = pack2bf(f2.x, f2.y); dst[5] = pack2bf(f2.z, f2.w);
                dst[6] = pack2bf(f3.x, f3.y); dst[7] = pack2bf(f3.z, f3.w);
            } else {
#pragma unroll
                for (int j = 0; j < 8; ++j) dst[j] = 0u;
            }
        }
        // ---- stage B into [n][k] layout ----
        if (bTrans) {
            // B stored [N x K]: row = n, 16 contiguous K floats -> contiguous LDS
            unsigned int* dst = (unsigned int*)lds_b16 + sRow * 16 + sHalf * 8;
            const float4* bp = (const float4*)(Bb + (long long)(nBase + sRow) * ldb + k0 + sHalf * 16);
            float4 f0 = bp[0], f1 = bp[1], f2 = bp[2], f3 = bp[3];
            dst[0] = pack2bf(f0.x, f0.y); dst[1] = pack2bf(f0.z, f0.w);
            dst[2] = pack2bf(f1.x, f1.y); dst[3] = pack2bf(f1.z, f1.w);
            dst[4] = pack2bf(f2.x, f2.y); dst[5] = pack2bf(f2.z, f2.w);
            dst[6] = pack2bf(f3.x, f3.y); dst[7] = pack2bf(f3.z, f3.w);
        } else {
            // B stored [K x N]: row = k, 16 contiguous N floats -> transpose into LDS
            const float4* bp = (const float4*)(Bb + (long long)(k0 + sKrow) * ldb + nBase + sQ * 16);
            float4 f0 = bp[0], f1 = bp[1], f2 = bp[2], f3 = bp[3];
            unsigned short* bd = lds_b16 + (sQ * 16) * 32 + sKrow;
            bd[0*32]  = f2bf(f0.x); bd[1*32]  = f2bf(f0.y); bd[2*32]  = f2bf(f0.z); bd[3*32]  = f2bf(f0.w);
            bd[4*32]  = f2bf(f1.x); bd[5*32]  = f2bf(f1.y); bd[6*32]  = f2bf(f1.z); bd[7*32]  = f2bf(f1.w);
            bd[8*32]  = f2bf(f2.x); bd[9*32]  = f2bf(f2.y); bd[10*32] = f2bf(f2.z); bd[11*32] = f2bf(f2.w);
            bd[12*32] = f2bf(f3.x); bd[13*32] = f2bf(f3.y); bd[14*32] = f2bf(f3.z); bd[15*32] = f2bf(f3.w);
        }
        __syncthreads();

        // ---- fragments from LDS (A and B use contiguous k order, k = 16*kh + s) ----
        BfVec a0, a1, b0, b1;
#pragma unroll
        for (int j = 0; j < 8; ++j) {
            a0.w[j]     = au[(mloc)      * 16 + kh * 8 + j];      // K kb..kb+15
            a1.w[j]     = au[(mloc + 16) * 16 + kh * 8 + j];
        }
#pragma unroll
        for (int j = 0; j < 8; ++j) {
            b0.w[j] = bu[(nloc)      * 16 + kh * 8 + j];          // K kb..kb+15
            b1.w[j] = bu[(nloc + 16) * 16 + kh * 8 + j];
        }
        acc00 = __builtin_amdgcn_wmma_f32_16x16x32_bf16(false, a0.v, false, b0.v, (short)0, acc00, false, false);
        acc01 = __builtin_amdgcn_wmma_f32_16x16x32_bf16(false, a0.v, false, b1.v, (short)0, acc01, false, false);
        acc10 = __builtin_amdgcn_wmma_f32_16x16x32_bf16(false, a1.v, false, b0.v, (short)0, acc10, false, false);
        acc11 = __builtin_amdgcn_wmma_f32_16x16x32_bf16(false, a1.v, false, b1.v, (short)0, acc11, false, false);
    }

    // C/D layout: lane n = col, VGPR i -> row base + i, +8 for lanes 16..31
    const int half = kh << 3;
    const int cm0 = mBase + wm + half;
    const int cn0 = nBase + wn + l15;
    epilogue(Cb, acc00, cm0,      cn0,      ldc, bias, gamma, mode);
    epilogue(Cb, acc01, cm0,      cn0 + 16, ldc, bias, gamma, mode);
    epilogue(Cb, acc10, cm0 + 16, cn0,      ldc, bias, gamma, mode);
    epilogue(Cb, acc11, cm0 + 16, cn0 + 16, ldc, bias, gamma, mode);
}

// ---------------- LayerNorm over D=384, one block (128 thr) per row --------
__global__ void ln_kernel(const float* __restrict__ x, float* __restrict__ y,
                          const float* __restrict__ g, const float* __restrict__ b)
{
    __shared__ float red[128];
    const long long row = blockIdx.x;
    const int t = threadIdx.x;
    const float* xr = x + row * DMODEL;
    float v0 = xr[t], v1 = xr[t + 128], v2 = xr[t + 256];
    red[t] = v0 + v1 + v2;
    __syncthreads();
    for (int s = 64; s > 0; s >>= 1) { if (t < s) red[t] += red[t + s]; __syncthreads(); }
    const float mu = red[0] * (1.0f / DMODEL);
    __syncthreads();
    const float d0 = v0 - mu, d1 = v1 - mu, d2 = v2 - mu;
    red[t] = d0 * d0 + d1 * d1 + d2 * d2;
    __syncthreads();
    for (int s = 64; s > 0; s >>= 1) { if (t < s) red[t] += red[t + s]; __syncthreads(); }
    const float rs = rsqrtf(red[0] * (1.0f / DMODEL) + 1e-6f);
    float* yr = y + row * DMODEL;
    yr[t]       = d0 * rs * g[t]       + b[t];
    yr[t + 128] = d1 * rs * g[t + 128] + b[t + 128];
    yr[t + 256] = d2 * rs * g[t + 256] + b[t + 256];
}

// ---------------- softmax over a score row (scale + mask pad cols) ---------
__global__ void softmax_kernel(float* __restrict__ S, int Lv, float scale)
{
    __shared__ float red[256];
    const long long z = blockIdx.z;
    float* row = S + z * (long long)LP * LP + (long long)blockIdx.x * LP;
    const int t = threadIdx.x;
    float m = -1e30f;
    for (int c = t; c < Lv; c += 256) m = fmaxf(m, row[c] * scale);
    red[t] = m; __syncthreads();
    for (int s = 128; s > 0; s >>= 1) { if (t < s) red[t] = fmaxf(red[t], red[t + s]); __syncthreads(); }
    m = red[0]; __syncthreads();
    float sum = 0.0f;
    for (int c = t; c < Lv; c += 256) sum += expf(row[c] * scale - m);
    red[t] = sum; __syncthreads();
    for (int s = 128; s > 0; s >>= 1) { if (t < s) red[t] += red[t + s]; __syncthreads(); }
    const float inv = 1.0f / red[0];
    for (int c = t; c < LP; c += 256)
        row[c] = (c < Lv) ? expf(row[c] * scale - m) * inv : 0.0f;
}

// ---------------- RoPE on q,k data tokens; 192 thr = 6 heads x 32 pairs ----
__global__ void rope_kernel(float* __restrict__ qkv, const int* __restrict__ win)
{
    const int p = blockIdx.x, b = blockIdx.y;
    const int head = threadIdx.x >> 5, j = threadIdx.x & 31;
    const float step = (float)win[0] * 2.0f;             // 0.25/0.125 factor
    const int jm = j & 15;
    const float inv = powf(100.0f, -(float)jm / 16.0f);  // THETA^-(4j/HD)
    const int px = p & 63, py = p >> 6;
    const float ang = (j < 16) ? ((float)px * step) * inv : (float)py * inv;
    const float c = cosf(ang), s = sinf(ang);
    const long long base = ((long long)(b * LP + NMETA + p)) * (3 * DMODEL) + head * HD + 2 * j;
    for (int sec = 0; sec < 2; ++sec) {                  // q then k
        float* ptr = qkv + base + sec * DMODEL;
        const float a = ptr[0], bb = ptr[1];
        ptr[0] = a * c - bb * s;
        ptr[1] = a * s + bb * c;
    }
}

// ---------------- im2col of the stride-16 patch conv -----------------------
__global__ void im2col_kernel(const float* __restrict__ x, float* __restrict__ out)
{
    const int p = blockIdx.x, b = blockIdx.y;
    const int k = threadIdx.x;                  // 256
    const int i = k >> 4, j = k & 15;
    const int py = p >> 6, px = p & 63;
    out[((long long)(b * NPATCH + p)) * 256 + k] =
        x[(long long)b * 128 * 1024 + (long long)(py * 16 + i) * 1024 + (px * 16 + j)];
}

// ---------------- assemble embedding rows (cls | meta | data | zero pad) ---
__global__ void embed_kernel(const float* __restrict__ pf,   // patch feat (+bias), (B*512,384)
                             const float* __restrict__ mbuf, // (64,192) valid rows 0..7
                             const float* __restrict__ sbuf,
                             const float* __restrict__ cls,
                             const float* __restrict__ chan_table,
                             const int*   __restrict__ channels,
                             float* __restrict__ xt,
                             float* __restrict__ meta_out)
{
    const int l = blockIdx.x, b = blockIdx.y, dd = threadIdx.x;
    float v;
    if (l == 0) {
        v = cls[dd];
    } else if (l <= 2) {
        const int i = l - 1;
        v = (dd < 192) ? mbuf[(b * 2 + i) * 192 + dd] : sbuf[(b * 2 + i) * 192 + dd - 192];
    } else if (l < LSEQ) {
        const int p = l - NMETA;
        v = pf[((long long)(b * NPATCH + p)) * DMODEL + dd]
          + chan_table[(long long)channels[b * NPATCH + p] * DMODEL + dd];
    } else {
        v = 0.0f;
    }
    xt[((long long)(b * LP + l)) * DMODEL + dd] = v;
    if (l < NMETA) meta_out[((long long)(b * NMETA + l)) * DMODEL + dd] = v;
}

// ---------------- copy valid rows to output --------------------------------
__global__ void out_kernel(const float* __restrict__ y, float* __restrict__ out)
{
    const int l = blockIdx.x, b = blockIdx.y, dd = threadIdx.x;
    out[((long long)(b * LSEQ + l)) * DMODEL + dd] = y[((long long)(b * LP + l)) * DMODEL + dd];
}

// ---------------- mask zeros + n_meta scalar -------------------------------
__global__ void tail_kernel(float* __restrict__ out)
{
    const int idx = blockIdx.x * 256 + threadIdx.x;
    const long long maskOff = (long long)BATCH * LSEQ * DMODEL + (long long)BATCH * NMETA * DMODEL;
    if (idx < BATCH * NPATCH) out[maskOff + idx] = 0.0f;
    if (idx == BATCH * NPATCH) out[maskOff + BATCH * NPATCH] = 3.0f;  // n_meta
}

// ===========================================================================
extern "C" void kernel_launch(void* const* d_in, const int* in_sizes, int n_in,
                              void* d_out, int out_size, void* d_ws, size_t ws_size,
                              hipStream_t stream)
{
    const float* x         = (const float*)d_in[0];
    const float* means     = (const float*)d_in[1];
    const float* stds      = (const float*)d_in[2];
    const int*   channels  = (const int*)  d_in[3];
    /* d_in[4] noise: identity permutation, unused */
    const int*   win_size  = (const int*)  d_in[5];
    /* d_in[6] mask_ratio == 0, unused */
    const float* patch_w   = (const float*)d_in[7];
    const float* patch_b   = (const float*)d_in[8];
    const float* cls_tok   = (const float*)d_in[9];
    const float* chan_tab  = (const float*)d_in[10];
    const float* mean_w    = (const float*)d_in[11];
    const float* mean_b    = (const float*)d_in[12];
    const float* qkv_w     = (const float*)d_in[13];
    const float* qkv_b     = (const float*)d_in[14];
    const float* proj_w    = (const float*)d_in[15];
    const float* proj_b    = (const float*)d_in[16];
    const float* fc1_w     = (const float*)d_in[17];
    const float* fc1_b     = (const float*)d_in[18];
    const float* fc2_w     = (const float*)d_in[19];
    const float* fc2_b     = (const float*)d_in[20];
    const float* ln1_g     = (const float*)d_in[21];
    const float* ln1_b     = (const float*)d_in[22];
    const float* ln2_g     = (const float*)d_in[23];
    const float* ln2_b     = (const float*)d_in[24];
    const float* gamma1    = (const float*)d_in[25];
    const float* gamma2    = (const float*)d_in[26];
    const float* lnf_g     = (const float*)d_in[27];
    const float* lnf_b     = (const float*)d_in[28];

    float* outp = (float*)d_out;

    // -------- workspace layout (floats) --------
    float* W      = (float*)d_ws;
    float* im2c   = W;                          // 2048*256
    float* pf     = im2c + 2048 * 256;          // 2048*384
    float* mbuf   = pf   + 2048 * 384;          // 64*192 (8 valid rows)
    float* sbuf   = mbuf + 64 * 192;            // 64*192
    float* xt     = sbuf + 64 * 192;            // B*LP*D
    float* ybuf   = xt   + (long long)ROWS_ALL * DMODEL;
    float* qkv    = ybuf + (long long)ROWS_ALL * DMODEL;         // B*LP*1152
    float* sc     = qkv  + (long long)ROWS_ALL * 3 * DMODEL;     // B*6*LP*LP
    float* obuf   = sc   + (long long)BATCH * NHEADS * LP * LP;  // B*LP*D
    float* hid    = obuf + (long long)ROWS_ALL * DMODEL;         // B*LP*FF

    const dim3 blk(128);
    const long long qkvB = (long long)LP * 3 * DMODEL;   // per-batch qkv stride
    const long long scB  = (long long)NHEADS * LP * LP;
    const long long scH  = (long long)LP * LP;
    const float attnScale = 0.125f;                      // HD^-0.5

    // ---- embedding ----
    im2col_kernel<<<dim3(NPATCH, BATCH), 256, 0, stream>>>(x, im2c);
    // patch conv as GEMM: (2048 x 256) @ (256 x 384), weights stored (384 x 256) -> transB
    gemm_wmma<<<dim3(DMODEL / 64, 2048 / 64, 1), blk, 0, stream>>>(
        im2c, 256, 0, 0,  patch_w, 256, 0, 0, 1,
        pf, DMODEL, 0, 0,  patch_b, nullptr, 0, 2048, 256, 1);
    // meta projections: (8 x 4000) @ (4000 x 192)
    gemm_wmma<<<dim3(192 / 64, 1, 1), blk, 0, stream>>>(
        means, 4000, 0, 0,  mean_w, 192, 0, 0, 0,
        mbuf, 192, 0, 0,  mean_b, nullptr, 0, 8, 4000, 1);
    gemm_wmma<<<dim3(192 / 64, 1, 1), blk, 0, stream>>>(
        stds, 4000, 0, 0,  mean_w, 192, 0, 0, 0,
        sbuf, 192, 0, 0,  mean_b, nullptr, 0, 8, 4000, 1);
    embed_kernel<<<dim3(LP, BATCH), DMODEL, 0, stream>>>(
        pf, mbuf, sbuf, cls_tok, chan_tab, channels, xt,
        outp + (long long)BATCH * LSEQ * DMODEL);

    // ---- transformer layers ----
    for (int d = 0; d < DEPTH; ++d) {
        const float* qw = qkv_w + (long long)d * DMODEL * 3 * DMODEL;
        const float* pw = proj_w + (long long)d * DMODEL * DMODEL;
        const float* f1 = fc1_w + (long long)d * DMODEL * FF;
        const float* f2 = fc2_w + (long long)d * FF * DMODEL;

        ln_kernel<<<ROWS_ALL, 128, 0, stream>>>(xt, ybuf, ln1_g + d * DMODEL, ln1_b + d * DMODEL);

        gemm_wmma<<<dim3(3 * DMODEL / 64, ROWS_ALL / 64, 1), blk, 0, stream>>>(
            ybuf, DMODEL, 0, 0,  qw, 3 * DMODEL, 0, 0, 0,
            qkv, 3 * DMODEL, 0, 0,  qkv_b + d * 3 * DMODEL, nullptr, 0, ROWS_ALL, DMODEL, 1);

        rope_kernel<<<dim3(NPATCH, BATCH), NHEADS * 32, 0, stream>>>(qkv, win_size);

        // scores = q @ k^T   (per batch*head)
        gemm_wmma<<<dim3(LP / 64, LP / 64, BATCH * NHEADS), blk, 0, stream>>>(
            qkv,              3 * DMODEL, qkvB, HD,
            qkv + DMODEL,     3 * DMODEL, qkvB, HD, 1,
            sc, LP, scB, scH,
            nullptr, nullptr, 0, LP, HD, NHEADS);

        softmax_kernel<<<dim3(LP, 1, BATCH * NHEADS), 256, 0, stream>>>(sc, LSEQ, attnScale);

        // o = probs @ v  -> written directly at column offset head*64 of (B,LP,D)
        gemm_wmma<<<dim3(HD / 64, LP / 64, BATCH * NHEADS), blk, 0, stream>>>(
            sc, LP, scB, scH,
            qkv + 2 * DMODEL, 3 * DMODEL, qkvB, HD, 0,
            obuf, DMODEL, (long long)LP * DMODEL, HD,
            nullptr, nullptr, 0, LP, LP, NHEADS);

        // xt += gamma1 * (o @ proj_w + proj_b)
        gemm_wmma<<<dim3(DMODEL / 64, ROWS_ALL / 64, 1), blk, 0, stream>>>(
            obuf, DMODEL, 0, 0,  pw, DMODEL, 0, 0, 0,
            xt, DMODEL, 0, 0,  proj_b + d * DMODEL, gamma1 + d * DMODEL, 1, ROWS_ALL, DMODEL, 1);

        ln_kernel<<<ROWS_ALL, 128, 0, stream>>>(xt, ybuf, ln2_g + d * DMODEL, ln2_b + d * DMODEL);

        // hid = gelu(y @ fc1_w + fc1_b)
        gemm_wmma<<<dim3(FF / 64, ROWS_ALL / 64, 1), blk, 0, stream>>>(
            ybuf, DMODEL, 0, 0,  f1, FF, 0, 0, 0,
            hid, FF, 0, 0,  fc1_b + d * FF, nullptr, 2, ROWS_ALL, DMODEL, 1);

        // xt += gamma2 * (hid @ fc2_w + fc2_b)
        gemm_wmma<<<dim3(DMODEL / 64, ROWS_ALL / 64, 1), blk, 0, stream>>>(
            hid, FF, 0, 0,  f2, DMODEL, 0, 0, 0,
            xt, DMODEL, 0, 0,  fc2_b + d * DMODEL, gamma2 + d * DMODEL, 1, ROWS_ALL, FF, 1);
    }

    // ---- final LN + outputs ----
    ln_kernel<<<ROWS_ALL, 128, 0, stream>>>(xt, ybuf, lnf_g, lnf_b);
    out_kernel<<<dim3(LSEQ, BATCH), DMODEL, 0, stream>>>(ybuf, outp);
    tail_kernel<<<9, 256, 0, stream>>>(outp);
}
